// Attention_72765335928968
// MI455X (gfx1250) — matrix operand order repaired and checked by reading, not compiled, not run on hardware
//
#include <hip/hip_runtime.h>
#include <hip/hip_bf16.h>

typedef __bf16 v16bf __attribute__((ext_vector_type(16)));
typedef __bf16 v8bf  __attribute__((ext_vector_type(8)));
typedef float  v8f   __attribute__((ext_vector_type(8)));

#define DIM      768
#define HEADS    12
#define HD       64
#define SEQ      1024
#define BATCH    8
#define OC       (3 * DIM)          // 2304
#define MROWS    (BATCH * SEQ)      // 8192
// Q pre-scale folds softmax's log2(e) so the inner loop uses raw exp2:
// exp(s*0.125 - m) == exp2(s*0.125*log2e - m')
#define QSCALE   (0.125f * 1.44269504088896341f)

#if __has_builtin(__builtin_amdgcn_global_load_async_to_lds_b128) && \
    __has_builtin(__builtin_amdgcn_s_wait_asynccnt)
#define ASYNC_OK 1
typedef int v4i __attribute__((__vector_size__(4 * sizeof(int))));
typedef __attribute__((address_space(1))) void* as1_void;
typedef __attribute__((address_space(3))) void* as3_void;
typedef __attribute__((address_space(1))) v4i*  as1_v4i;
typedef __attribute__((address_space(3))) v4i*  as3_v4i;

static __device__ __forceinline__ void async_lds_b128x4(const void* g, void* l) {
    as1_v4i gp = (as1_v4i)(as1_void)(void*)g;
    as3_v4i lp = (as3_v4i)(as3_void)l;
    __builtin_amdgcn_global_load_async_to_lds_b128(gp, lp, 0,  0);
    __builtin_amdgcn_global_load_async_to_lds_b128(gp, lp, 16, 0);
    __builtin_amdgcn_global_load_async_to_lds_b128(gp, lp, 32, 0);
    __builtin_amdgcn_global_load_async_to_lds_b128(gp, lp, 48, 0);
}
#else
#define ASYNC_OK 0
#endif

static __device__ __forceinline__ v8f wmma_bf16(v16bf a, v16bf b, v8f c) {
    // D(16x16 f32) = A(16x32 bf16) * B(32x16 bf16) + C
    return __builtin_amdgcn_wmma_f32_16x16x32_bf16(
        /*neg_a=*/false, a, /*neg_b=*/false, b,
        /*c_mod=*/(short)0, c, /*reuse_a=*/false, /*reuse_b=*/false);
}

static __device__ __forceinline__ v8f zero8() {
    v8f z = {0.f,0.f,0.f,0.f,0.f,0.f,0.f,0.f};
    return z;
}

// ---------------------------------------------------------------------------
// Kernel 0: elementwise f32 -> bf16 (8 elems/thread). Run once per call for
// x, qkv_w, proj_w so GEMM inner loops are pure b128-load -> WMMA (no cvt).
// ---------------------------------------------------------------------------
__global__ __launch_bounds__(256)
void cvt_bf16_kernel(const float* __restrict__ in, __bf16* __restrict__ out,
                     int n8)
{
    const int i = blockIdx.x * 256 + threadIdx.x;
    if (i >= n8) return;
    const float4 f0 = ((const float4*)in)[i * 2];
    const float4 f1 = ((const float4*)in)[i * 2 + 1];
    v8bf o;
    o[0] = (__bf16)f0.x; o[1] = (__bf16)f0.y;
    o[2] = (__bf16)f0.z; o[3] = (__bf16)f0.w;
    o[4] = (__bf16)f1.x; o[5] = (__bf16)f1.y;
    o[6] = (__bf16)f1.z; o[7] = (__bf16)f1.w;
    ((v8bf*)out)[i] = o;
}

// ---------------------------------------------------------------------------
// Kernel 1: qkv = x @ qkv_w^T (both bf16) -> split into Q (pre-scaled, bf16),
// K, V, each [B, H, N, 64]. 256 threads = 8 waves; block tile 128x128;
// wave tile 32x64 (2x4 accum tiles), K-step 32.
// ---------------------------------------------------------------------------
__global__ __launch_bounds__(256)
void qkv_kernel(const __bf16* __restrict__ x, const __bf16* __restrict__ w,
                __bf16* __restrict__ Qo, __bf16* __restrict__ Ko,
                __bf16* __restrict__ Vo)
{
    const int wave = threadIdx.x >> 5;
    const int lane = threadIdx.x & 31;
    const int hi = lane >> 4, lo = lane & 15;
    const int rowBase = blockIdx.x * 128 + (wave >> 1) * 32;
    const int colBase = blockIdx.y * 128 + (wave & 1) * 64;

    v8f acc[2][4];
    #pragma unroll
    for (int mt = 0; mt < 2; ++mt)
        #pragma unroll
        for (int nt = 0; nt < 4; ++nt) acc[mt][nt] = zero8();

    for (int kk = 0; kk < DIM; kk += 32) {
        v16bf afrag[2];
        #pragma unroll
        for (int mt = 0; mt < 2; ++mt) {
            const __bf16* ap = x + (size_t)(rowBase + mt * 16 + lo) * DIM + kk;
            v8bf h0 = *(const v8bf*)(ap + hi * 16);
            v8bf h1 = *(const v8bf*)(ap + hi * 16 + 8);
            #pragma unroll
            for (int j = 0; j < 8; ++j) {
                afrag[mt][j] = h0[j];
                afrag[mt][j + 8] = h1[j];
            }
        }
        #pragma unroll
        for (int nt = 0; nt < 4; ++nt) {
            const v16bf bfrag = *(const v16bf*)
                (w + (size_t)(colBase + nt * 16 + lo) * DIM + kk + hi * 16);
            #pragma unroll
            for (int mt = 0; mt < 2; ++mt)
                acc[mt][nt] = wmma_bf16(afrag[mt], bfrag, acc[mt][nt]);
        }
    }

    // Epilogue: scatter into Q/K/V [B,H,N,64] bf16 (Q pre-scaled w/ log2e).
    #pragma unroll
    for (int mt = 0; mt < 2; ++mt) {
        #pragma unroll
        for (int nt = 0; nt < 4; ++nt) {
            const int col   = colBase + nt * 16 + lo;   // 0..2303
            const int which = col / DIM;                // 0=q 1=k 2=v
            const int rem   = col % DIM;
            const int h     = rem / HD;
            const int d     = rem % HD;
            #pragma unroll
            for (int r = 0; r < 8; ++r) {
                const int row = rowBase + mt * 16 + r + 8 * hi;  // 0..8191
                const int b = row >> 10, n = row & 1023;
                const size_t idx = (((size_t)(b * HEADS + h) * SEQ) + n) * HD + d;
                const float v = acc[mt][nt][r];
                if (which == 0)      Qo[idx] = (__bf16)(v * QSCALE);
                else if (which == 1) Ko[idx] = (__bf16)v;
                else                 Vo[idx] = (__bf16)v;
            }
        }
    }
}

// ---------------------------------------------------------------------------
// Kernel 2: flash attention per (b,h, 64-query block). 128 threads = 4 waves;
// wave owns 16 query rows. K tiles double-buffered into LDS via async
// load-to-LDS (ASYNCcnt pipelined); V staged transposed; online softmax in
// exp2 domain. Output -> AO [B, N, 768] bf16.
// ---------------------------------------------------------------------------
__global__ __launch_bounds__(128)
void attn_kernel(const __bf16* __restrict__ Q, const __bf16* __restrict__ K,
                 const __bf16* __restrict__ V, __bf16* __restrict__ AO)
{
    __shared__ __bf16 Kl[2][64][64];     // K tile double buffer (row-major)
    __shared__ __bf16 Vt[HD][80];        // V transposed: Vt[d][key_local]
    __shared__ __bf16 Pl[4][16][64];     // per-wave probability tile

    const int tid  = threadIdx.x;
    const int wave = tid >> 5;
    const int lane = tid & 31;
    const int hi = lane >> 4, lo = lane & 15;

    const int qb = (blockIdx.x & 15) * 64;   // query block base
    const int bh = blockIdx.x >> 4;          // b*HEADS + h
    const __bf16* Qp = Q + (size_t)bh * SEQ * HD;
    const __bf16* Kp = K + (size_t)bh * SEQ * HD;
    const __bf16* Vp = V + (size_t)bh * SEQ * HD;

    // Q A-fragments for this wave's 16 rows, held for the whole key loop.
    v16bf qfrag[2];
    #pragma unroll
    for (int ks = 0; ks < 2; ++ks) {
        const __bf16* p = Qp + (size_t)(qb + wave * 16 + lo) * HD + ks * 32;
        #pragma unroll
        for (int j = 0; j < 8; ++j) {
            qfrag[ks][j]     = p[hi * 16 + j];
            qfrag[ks][j + 8] = p[hi * 16 + 8 + j];
        }
    }

    float rowmax[8], rowsum[8];
    v8f O[4];
    #pragma unroll
    for (int r = 0; r < 8; ++r) { rowmax[r] = -INFINITY; rowsum[r] = 0.f; }
    #pragma unroll
    for (int ot = 0; ot < 4; ++ot) O[ot] = zero8();

#if ASYNC_OK
    // prologue: async-load K tile 0 into buffer 0 (64B per thread)
    async_lds_b128x4(Kp + (size_t)tid * 32, &Kl[0][0][0] + tid * 32);
#endif

    for (int kb = 0; kb < SEQ / 64; ++kb) {
        const int keyBase = kb * 64;
        const int buf = kb & 1;
        if (kb + 1 < SEQ / 64)
            __builtin_prefetch(Vp + (size_t)(keyBase + 64) * HD, 0, 1);

        __syncthreads();   // Vt / Kl WAR vs previous iteration's readers

#if ASYNC_OK
        if (kb + 1 < SEQ / 64)     // async-load next K tile into other buffer
            async_lds_b128x4(Kp + (size_t)(keyBase + 64) * HD + tid * 32,
                             &Kl[buf ^ 1][0][0] + tid * 32);
#else
        {   // fallback: synchronous K tile stage
            const __bf16* g = Kp + (size_t)keyBase * HD + tid * 32;
            __bf16* l = &Kl[buf][0][0] + tid * 32;
            #pragma unroll
            for (int j = 0; j < 32; ++j) l[j] = g[j];
        }
#endif
        {   // cooperative stage of V^T into LDS (64x64 bf16)
            const int kr = tid >> 1, half = tid & 1;
            const __bf16* vp = Vp + (size_t)(keyBase + kr) * HD + half * 32;
            #pragma unroll
            for (int j = 0; j < 32; ++j) Vt[half * 32 + j][kr] = vp[j];
        }
#if ASYNC_OK
        // Async ops complete in order: waiting <=4 retires this wave's 4 ops
        // for tile kb, leaving kb+1's 4 in flight.
        if (kb + 1 < SEQ / 64) __builtin_amdgcn_s_wait_asynccnt(4);
        else                   __builtin_amdgcn_s_wait_asynccnt(0);
#endif
        __syncthreads();

        // S = (Q*scale*log2e) @ K^T : 4 tiles of 16x16, reduce over d=64.
        v8f S[4];
        #pragma unroll
        for (int nt = 0; nt < 4; ++nt) {
            v8f acc = zero8();
            #pragma unroll
            for (int ks = 0; ks < 2; ++ks) {
                const v16bf bfrag =
                    *(const v16bf*)&Kl[buf][nt * 16 + lo][ks * 32 + hi * 16];
                acc = wmma_bf16(qfrag[ks], bfrag, acc);
            }
            S[nt] = acc;
        }

        // Online softmax (exp2 domain). Lane holds column lo, rows r + 8*hi.
        #pragma unroll
        for (int r = 0; r < 8; ++r) {
            float t = fmaxf(fmaxf(S[0][r], S[1][r]), fmaxf(S[2][r], S[3][r]));
            #pragma unroll
            for (int off = 8; off >= 1; off >>= 1)
                t = fmaxf(t, __shfl_xor(t, off, 32));
            const float nm    = fmaxf(rowmax[r], t);
            const float alpha = exp2f(rowmax[r] - nm);
            rowmax[r] = nm;
            float ps = 0.f;
            #pragma unroll
            for (int nt = 0; nt < 4; ++nt) {
                const float p = exp2f(S[nt][r] - nm);
                S[nt][r] = p;
                ps += p;
            }
            #pragma unroll
            for (int off = 8; off >= 1; off >>= 1)
                ps += __shfl_xor(ps, off, 32);
            rowsum[r] = rowsum[r] * alpha + ps;
            #pragma unroll
            for (int ot = 0; ot < 4; ++ot) O[ot][r] *= alpha;
        }

        // P: C-layout -> LDS -> A-layout (per-wave buffer, same-wave DS order).
        #pragma unroll
        for (int nt = 0; nt < 4; ++nt)
            #pragma unroll
            for (int r = 0; r < 8; ++r)
                Pl[wave][r + 8 * hi][nt * 16 + lo] = (__bf16)S[nt][r];

        // O += P @ V : 8 WMMAs (2 K-steps x 4 d-subtiles).
        #pragma unroll
        for (int ks = 0; ks < 2; ++ks) {
            const __bf16* pp = &Pl[wave][lo][ks * 32 + hi * 16];
            v16bf pf;
            #pragma unroll
            for (int j = 0; j < 8; ++j) { pf[j] = pp[j]; pf[j + 8] = pp[8 + j]; }
            #pragma unroll
            for (int ot = 0; ot < 4; ++ot) {
                const __bf16* vv = &Vt[ot * 16 + lo][ks * 32 + hi * 16];
                v16bf vf;
                #pragma unroll
                for (int j = 0; j < 16; ++j) vf[j] = vv[j];
                O[ot] = wmma_bf16(pf, vf, O[ot]);
            }
        }
    }

    // Normalize + write [B, N, 768] bf16 (head h at column h*64).
    const int b = bh / HEADS, h = bh % HEADS;
    #pragma unroll
    for (int r = 0; r < 8; ++r) {
        const float inv = 1.f / rowsum[r];
        const int q = qb + wave * 16 + r + 8 * hi;
        __bf16* op = AO + ((size_t)b * SEQ + q) * DIM + h * HD;
        #pragma unroll
        for (int ot = 0; ot < 4; ++ot)
            op[ot * 16 + lo] = (__bf16)(O[ot][r] * inv);
    }
}

// ---------------------------------------------------------------------------
// Kernel 3: out = AO @ proj_w^T + proj_b (f32 output). Same tiling as qkv.
// ---------------------------------------------------------------------------
__global__ __launch_bounds__(256)
void proj_kernel(const __bf16* __restrict__ ao, const __bf16* __restrict__ w,
                 const float* __restrict__ bias, float* __restrict__ out)
{
    const int wave = threadIdx.x >> 5;
    const int lane = threadIdx.x & 31;
    const int hi = lane >> 4, lo = lane & 15;
    const int rowBase = blockIdx.x * 128 + (wave >> 1) * 32;
    const int colBase = blockIdx.y * 128 + (wave & 1) * 64;

    v8f acc[2][4];
    #pragma unroll
    for (int mt = 0; mt < 2; ++mt)
        #pragma unroll
        for (int nt = 0; nt < 4; ++nt) acc[mt][nt] = zero8();

    for (int kk = 0; kk < DIM; kk += 32) {
        v16bf afrag[2];
        #pragma unroll
        for (int mt = 0; mt < 2; ++mt) {
            const __bf16* ap = ao + (size_t)(rowBase + mt * 16 + lo) * DIM + kk;
            v8bf h0 = *(const v8bf*)(ap + hi * 16);
            v8bf h1 = *(const v8bf*)(ap + hi * 16 + 8);
            #pragma unroll
            for (int j = 0; j < 8; ++j) {
                afrag[mt][j] = h0[j];
                afrag[mt][j + 8] = h1[j];
            }
        }
        #pragma unroll
        for (int nt = 0; nt < 4; ++nt) {
            const v16bf bfrag = *(const v16bf*)
                (w + (size_t)(colBase + nt * 16 + lo) * DIM + kk + hi * 16);
            #pragma unroll
            for (int mt = 0; mt < 2; ++mt)
                acc[mt][nt] = wmma_bf16(afrag[mt], bfrag, acc[mt][nt]);
        }
    }

    #pragma unroll
    for (int mt = 0; mt < 2; ++mt)
        #pragma unroll
        for (int nt = 0; nt < 4; ++nt) {
            const int col = colBase + nt * 16 + lo;
            const float bv = bias[col];
            #pragma unroll
            for (int r = 0; r < 8; ++r) {
                const int row = rowBase + mt * 16 + r + 8 * hi;
                out[(size_t)row * DIM + col] = acc[mt][nt][r] + bv;
            }
        }
}

// ---------------------------------------------------------------------------
extern "C" void kernel_launch(void* const* d_in, const int* in_sizes, int n_in,
                              void* d_out, int out_size, void* d_ws, size_t ws_size,
                              hipStream_t stream)
{
    const float* x      = (const float*)d_in[0];
    const float* qkv_w  = (const float*)d_in[1];
    const float* proj_w = (const float*)d_in[2];
    const float* proj_b = (const float*)d_in[3];
    float* out = (float*)d_out;

    const size_t elems   = (size_t)BATCH * HEADS * SEQ * HD;   // 6,291,456
    const size_t n_x     = (size_t)MROWS * DIM;                // 6,291,456
    const size_t n_wq    = (size_t)OC * DIM;                   // 1,769,472
    const size_t n_wp    = (size_t)DIM * DIM;                  //   589,824

    __bf16* Qs  = (__bf16*)d_ws;
    __bf16* Ks  = Qs + elems;
    __bf16* Vs  = Ks + elems;
    __bf16* AO  = Vs + elems;           // [B, N, 768] bf16
    __bf16* xb  = AO + elems;
    __bf16* wqb = xb + n_x;
    __bf16* wpb = wqb + n_wq;

    // 0) one-shot f32 -> bf16 conversion of x and both weight matrices
    cvt_bf16_kernel<<<dim3((int)(n_x  / 8 / 256)), 256, 0, stream>>>(x,      xb,  (int)(n_x  / 8));
    cvt_bf16_kernel<<<dim3((int)(n_wq / 8 / 256)), 256, 0, stream>>>(qkv_w,  wqb, (int)(n_wq / 8));
    cvt_bf16_kernel<<<dim3((int)(n_wp / 8 / 256)), 256, 0, stream>>>(proj_w, wpb, (int)(n_wp / 8));

    // 1) QKV projection (pure bf16 WMMA GEMM)
    qkv_kernel<<<dim3(MROWS / 128, OC / 128), 256, 0, stream>>>(
        xb, wqb, Qs, Ks, Vs);

    // 2) fused flash attention per (b, h, 64-query block)
    attn_kernel<<<dim3(BATCH * HEADS * (SEQ / 64)), 128, 0, stream>>>(
        Qs, Ks, Vs, AO);

    // 3) output projection + bias
    proj_kernel<<<dim3(MROWS / 128, DIM / 128), 256, 0, stream>>>(
        AO, wpb, proj_b, out);
}
